// TopMultiAttention_29197187678750
// MI455X (gfx1250) — matrix operand order repaired and checked by reading, not compiled, not run on hardware
//
#include <hip/hip_runtime.h>
#include <stdint.h>

// ---------------------------------------------------------------- types
typedef __bf16 bf16;
typedef __attribute__((ext_vector_type(16))) __bf16 v16bf;
typedef __attribute__((ext_vector_type(8)))  __bf16 v8bf;
typedef __attribute__((ext_vector_type(8)))  float  v8f;
typedef __attribute__((ext_vector_type(4)))  unsigned int v4u;
typedef __attribute__((ext_vector_type(8)))  int   v8i;
typedef __attribute__((ext_vector_type(4)))  int   v4i;

#define B_    4
#define S_    2048
#define D_    512
#define H_    8
#define HD_   64
#define MTOT  (B_ * S_)   // 8192

#if defined(__has_builtin)
#if __has_builtin(__builtin_amdgcn_tensor_load_to_lds)
#define HAVE_TDM5 1
#endif
#endif

// ---------------------------------------------------------------- helpers
__device__ __forceinline__ bf16 f2bf(float f) {
  uint32_t u = __builtin_bit_cast(uint32_t, f);
  u += 0x7FFFu + ((u >> 16) & 1u);       // round-to-nearest-even
  uint16_t h = (uint16_t)(u >> 16);
  return __builtin_bit_cast(bf16, h);
}
__device__ __forceinline__ float bf2f(bf16 h) {
  uint32_t u = ((uint32_t)__builtin_bit_cast(uint16_t, h)) << 16;
  return __builtin_bit_cast(float, u);
}
__device__ __forceinline__ float toF(float x) { return x; }
__device__ __forceinline__ float toF(bf16 x)  { return bf2f(x); }

__device__ __forceinline__ v16bf cat8(v8bf lo, v8bf hi) {
  return __builtin_shufflevector(lo, hi, 0,1,2,3,4,5,6,7,8,9,10,11,12,13,14,15);
}

__device__ __forceinline__ void wait_tensorcnt0() {
#if defined(__has_builtin) && __has_builtin(__builtin_amdgcn_s_wait_tensorcnt)
  __builtin_amdgcn_s_wait_tensorcnt(0);
#else
  asm volatile("s_wait_tensorcnt 0x0" ::: "memory");
#endif
}

// =================================================================
// GEMM: C[M,N] = A[M,K] * W[K,N]; A is f32 or bf16, W is f32.
// Output bf16 (projections) or f32 (final). M%128==0, N%128==0, K%32==0.
// Workgroup: 256 threads (8 waves), tile 128x128. Wave tile 32x64 (2x4 WMMA).
// =================================================================
template <typename AT, bool OUT_BF16>
__global__ __launch_bounds__(256)
void gemm_kernel(const AT* __restrict__ A, const float* __restrict__ W,
                 void* __restrict__ Cout, int M, int N, int K) {
  __shared__ __align__(16) bf16 As[128][40];   // [m][k], stride 80B (16B mult)
  __shared__ __align__(16) bf16 Ws[128][40];   // transposed: [n][k]

  const int tid  = threadIdx.x;
  const int lane = tid & 31;
  const int wave = tid >> 5;
  const int wm   = wave >> 1;            // 0..3  (32-row block)
  const int wn   = wave & 1;             // 0..1  (64-col block)
  const int m0   = blockIdx.x * 128;
  const int n0   = blockIdx.y * 128;
  const int hl   = lane >> 4;            // lane half
  const int l16  = lane & 15;

  v8f zero = {};
  v8f acc[2][4];
#pragma unroll
  for (int i = 0; i < 2; ++i)
#pragma unroll
    for (int j = 0; j < 4; ++j) acc[i][j] = zero;

  for (int k0 = 0; k0 < K; k0 += 32) {
    __syncthreads();
    // ---- stage A tile 128x32 (convert to bf16)
#pragma unroll
    for (int idx = tid; idx < 128 * 32; idx += 256) {
      int r = idx >> 5, c = idx & 31;
      As[r][c] = f2bf(toF(A[(size_t)(m0 + r) * K + (k0 + c)]));
    }
    // ---- stage W tile 32x128 transposed -> Ws[n][k]
#pragma unroll
    for (int idx = tid; idx < 128 * 32; idx += 256) {
      int k = idx >> 7, n = idx & 127;
      Ws[n][k] = f2bf(W[(size_t)(k0 + k) * N + (n0 + n)]);
    }
    // ---- prefetch next K-step tiles into cache (global_prefetch_b8)
    if (k0 + 32 < K) {
      __builtin_prefetch(&A[(size_t)(m0 + (tid >> 1)) * K + (k0 + 32) + (tid & 1) * 16], 0, 1);
      __builtin_prefetch(&W[(size_t)(k0 + 32 + (tid >> 3)) * N + n0 + (tid & 7) * 16], 0, 1);
    }
    __syncthreads();

    // ---- fragments
    v16bf afr[2];
#pragma unroll
    for (int mi = 0; mi < 2; ++mi) {
      int row = wm * 32 + mi * 16 + l16;
      int kb  = hl * 8;                              // A: K 0..7 | 8..15
      v8bf lo = *(const v8bf*)&As[row][kb];
      v8bf hi = *(const v8bf*)&As[row][kb + 16];     // K +16
      afr[mi] = cat8(lo, hi);
    }
    v16bf bfr[4];
#pragma unroll
    for (int ni = 0; ni < 4; ++ni) {
      int col = wn * 64 + ni * 16 + l16;
      int kb  = hl * 8;                              // B: K 0..7 | 8..15
      v8bf lo = *(const v8bf*)&Ws[col][kb];
      v8bf hi = *(const v8bf*)&Ws[col][kb + 16];     // K +16
      bfr[ni] = cat8(lo, hi);
    }
#pragma unroll
    for (int mi = 0; mi < 2; ++mi)
#pragma unroll
      for (int ni = 0; ni < 4; ++ni)
        acc[mi][ni] = __builtin_amdgcn_wmma_f32_16x16x32_bf16(
            false, afr[mi], false, bfr[ni], (short)0, acc[mi][ni], false, false);
  }

  // ---- store D (vgpr r -> row r + 8*half, col = lane%16)
#pragma unroll
  for (int mi = 0; mi < 2; ++mi)
#pragma unroll
    for (int ni = 0; ni < 4; ++ni)
#pragma unroll
      for (int r = 0; r < 8; ++r) {
        int gm = m0 + wm * 32 + mi * 16 + r + 8 * hl;
        int gn = n0 + wn * 64 + ni * 16 + l16;
        float v = acc[mi][ni][r];
        if (OUT_BF16) ((bf16*)Cout)[(size_t)gm * N + gn] = f2bf(v);
        else          ((float*)Cout)[(size_t)gm * N + gn] = v;
      }
}

// =================================================================
// Flash attention over relu(scores), per (b, h, 64-row q-tile).
// 128 threads = 4 waves; wave w owns q rows [16w, 16w+16).
// scores = (Q/8)·K^T ; P = exp(relu(s) - m) online; O = P·V ; O /= l.
// K tile staged by the Tensor Data Mover (2D D# with LDS row padding);
// V tile staged manually (needs a transpose for the WMMA B-fragment).
// =================================================================
__global__ __launch_bounds__(128)
void attn_kernel(const bf16* __restrict__ Qb, const bf16* __restrict__ Kb,
                 const bf16* __restrict__ Vb, bf16* __restrict__ Ob) {
  const int bh = blockIdx.x;             // 0..31
  const int qt = blockIdx.y;             // 0..31
  const int b  = bh / H_;
  const int h  = bh % H_;
  const int q0 = qt * 64;

  const size_t baseQ  = ((size_t)b * S_ + q0) * D_ + (size_t)h * HD_;
  const size_t baseKV = ((size_t)b * S_) * D_ + (size_t)h * HD_;

  __shared__ __align__(16) bf16 Qs[64][72];  // [q][d]  stride 144B (16B mult)
  __shared__ __align__(16) bf16 Ks[64][72];  // [kc][d] (TDM dest, 16B row pad)
  __shared__ __align__(16) bf16 Vt[64][72];  // [d][kc] (transposed)
  __shared__ __align__(16) bf16 Ps[64][72];  // [q][kc] probs

  const int tid  = threadIdx.x;
  const int lane = tid & 31;
  const int wave = tid >> 5;             // q-row block
  const int hl   = lane >> 4;
  const int l16  = lane & 15;

  // stage Q, pre-scaled by 1/sqrt(HD) = 0.125
  for (int idx = tid; idx < 64 * 64; idx += 128) {
    int r = idx >> 6, c = idx & 63;
    Qs[r][c] = f2bf(bf2f(Qb[baseQ + (size_t)r * D_ + c]) * 0.125f);
  }

  v8f zero = {};
  v8f o[4];                               // O accum: 4 d-tiles of 16x16
#pragma unroll
  for (int i = 0; i < 4; ++i) o[i] = zero;
  float m_cur[8], l_cur[8];               // per-lane rows r + 8*hl
#pragma unroll
  for (int r = 0; r < 8; ++r) { m_cur[r] = 0.f; l_cur[r] = 0.f; }

  for (int kc0 = 0; kc0 < S_; kc0 += 64) {
    __syncthreads();                      // protect Ks/Vt/Ps from prior iter

#ifdef HAVE_TDM5
    // ---- K tile 64x64 bf16 via Tensor Data Mover (wave 0 issues the DMA).
    // Tile row = 64 elem * 2B = 128B = 32 DWORDs -> pad_interval=4 (32 DW);
    // pad 16B (4 DW -> pad_amount=3) per row => LDS row stride 144B = Ks[..][72].
    if (wave == 0) {
      uint64_t ga   = (uint64_t)(uintptr_t)(Kb + baseKV + (size_t)kc0 * D_);
      uint32_t ldsa = (uint32_t)(uintptr_t)&Ks[0][0];
      v4u g0;
      g0[0] = 1u;                                   // count=1, no gather
      g0[1] = ldsa;                                 // lds_addr
      g0[2] = (uint32_t)ga;                         // global_addr[31:0]
      g0[3] = (uint32_t)(((ga >> 32) & 0x1FFFFFFu) | (2u << 30)); // [56:32]|type=2
      v8i g1;
      g1[0] = (int)((1u << 16) | (1u << 20) | (4u << 22) | (3u << 25));
              // data_size=2B | pad_enable | pad_interval=32DW | pad_amount=4DW
      g1[1] = (int)((uint32_t)D_ << 16);            // tensor_dim0 = 512
      g1[2] = (int)((uint32_t)S_ << 16);            // tensor_dim1 = 2048
      g1[3] = (int)(64u << 16);                     // tile_dim0 = 64
      g1[4] = 64;                                   // tile_dim1 = 64 (tile_dim2=0)
      g1[5] = D_;                                   // tensor_dim0_stride = 512
      g1[6] = 0;
      g1[7] = 0;
      v4i gz4 = {0, 0, 0, 0};
      v8i gz8 = {0, 0, 0, 0, 0, 0, 0, 0};
      __builtin_amdgcn_tensor_load_to_lds(g0, g1, gz4, gz4, gz8, 0);
    }
#endif

    // ---- V tile staged + transposed by all waves (co-executes with TDM)
    for (int idx = tid; idx < 64 * 64; idx += 128) {
      int r = idx >> 6, c = idx & 63;
#ifndef HAVE_TDM5
      Ks[r][c] = Kb[baseKV + (size_t)(kc0 + r) * D_ + c];
#endif
      Vt[c][r] = Vb[baseKV + (size_t)(kc0 + r) * D_ + c];
    }

#ifdef HAVE_TDM5
    if (wave == 0) wait_tensorcnt0();     // TDM done before barrier publishes Ks
#endif
    __syncthreads();

    // ---- S = Q · K^T  (wave's 16 q-rows x 64 kc), K-dim = 64 -> 2 steps
    v8f sacc[4];
#pragma unroll
    for (int i = 0; i < 4; ++i) sacc[i] = zero;
#pragma unroll
    for (int ks = 0; ks < 2; ++ks) {
      int row = wave * 16 + l16;
      int kb  = ks * 32 + hl * 8;
      v16bf af = cat8(*(const v8bf*)&Qs[row][kb],
                      *(const v8bf*)&Qs[row][kb + 16]);
#pragma unroll
      for (int ni = 0; ni < 4; ++ni) {
        int col = ni * 16 + l16;          // kc column
        int kb2 = ks * 32 + hl * 8;       // d (contraction)
        v16bf bf = cat8(*(const v8bf*)&Ks[col][kb2],
                        *(const v8bf*)&Ks[col][kb2 + 16]);
        sacc[ni] = __builtin_amdgcn_wmma_f32_16x16x32_bf16(
            false, af, false, bf, (short)0, sacc[ni], false, false);
      }
    }

    // ---- relu + per-row max (row = r + 8*hl within wave block)
    float tmax[8];
#pragma unroll
    for (int r = 0; r < 8; ++r) {
      float mx = 0.f;
#pragma unroll
      for (int ni = 0; ni < 4; ++ni) {
        float s = sacc[ni][r];
        s = s > 0.f ? s : 0.f;
        sacc[ni][r] = s;
        mx = fmaxf(mx, s);
      }
      tmax[r] = mx;
    }
#pragma unroll
    for (int off = 1; off < 16; off <<= 1)
#pragma unroll
      for (int r = 0; r < 8; ++r)
        tmax[r] = fmaxf(tmax[r], __shfl_xor(tmax[r], off, 32));

    float scl[8];
#pragma unroll
    for (int r = 0; r < 8; ++r) {
      float mn = fmaxf(m_cur[r], tmax[r]);
      scl[r]   = __expf(m_cur[r] - mn);
      m_cur[r] = mn;
    }

    // ---- P = exp(s - m); row sums; stage P to LDS (bf16)
    float rsum[8];
#pragma unroll
    for (int r = 0; r < 8; ++r) rsum[r] = 0.f;
#pragma unroll
    for (int ni = 0; ni < 4; ++ni)
#pragma unroll
      for (int r = 0; r < 8; ++r) {
        float p = __expf(sacc[ni][r] - m_cur[r]);
        rsum[r] += p;
        Ps[wave * 16 + r + 8 * hl][ni * 16 + l16] = f2bf(p);
      }
#pragma unroll
    for (int off = 1; off < 16; off <<= 1)
#pragma unroll
      for (int r = 0; r < 8; ++r)
        rsum[r] += __shfl_xor(rsum[r], off, 32);
#pragma unroll
    for (int r = 0; r < 8; ++r) l_cur[r] = l_cur[r] * scl[r] + rsum[r];

    // rescale running O
#pragma unroll
    for (int ni = 0; ni < 4; ++ni)
#pragma unroll
      for (int r = 0; r < 8; ++r) o[ni][r] *= scl[r];

    __syncthreads();                      // Ps visible

    // ---- O += P · V  (K-dim = 64 kc -> 2 steps)
#pragma unroll
    for (int ks = 0; ks < 2; ++ks) {
      int row = wave * 16 + l16;
      int kb  = ks * 32 + hl * 8;
      v16bf af = cat8(*(const v8bf*)&Ps[row][kb],
                      *(const v8bf*)&Ps[row][kb + 16]);
#pragma unroll
      for (int ni = 0; ni < 4; ++ni) {
        int col = ni * 16 + l16;          // d column
        int kb2 = ks * 32 + hl * 8;       // kc (contraction)
        v16bf bf = cat8(*(const v8bf*)&Vt[col][kb2],
                        *(const v8bf*)&Vt[col][kb2 + 16]);
        o[ni] = __builtin_amdgcn_wmma_f32_16x16x32_bf16(
            false, af, false, bf, (short)0, o[ni], false, false);
      }
    }
  }

  // ---- finalize: O /= l, store bf16 [B,S,D] merged-head layout
#pragma unroll
  for (int ni = 0; ni < 4; ++ni)
#pragma unroll
    for (int r = 0; r < 8; ++r) {
      int gm = q0 + wave * 16 + r + 8 * hl;
      int gd = ni * 16 + l16;
      float v = o[ni][r] / l_cur[r];
      Ob[((size_t)b * S_ + gm) * D_ + (size_t)h * HD_ + gd] = f2bf(v);
    }
}

// =================================================================
extern "C" void kernel_launch(void* const* d_in, const int* in_sizes, int n_in,
                              void* d_out, int out_size, void* d_ws, size_t ws_size,
                              hipStream_t stream) {
  (void)in_sizes; (void)n_in; (void)out_size; (void)ws_size;
  const float* X  = (const float*)d_in[0];
  const float* Y  = (const float*)d_in[1];
  const float* Z  = (const float*)d_in[2];
  const float* Wq = (const float*)d_in[3];
  const float* Wk = (const float*)d_in[4];
  // d_in[5] = Wv : dead in the reference forward
  const float* Wo = (const float*)d_in[6];
  float* out = (float*)d_out;

  const size_t elems = (size_t)MTOT * D_;
  bf16* Qb = (bf16*)d_ws;
  bf16* Kb = Qb + elems;
  bf16* Vb = Kb + elems;
  bf16* Ab = Vb + elems;    // attention output (pre-Wo), bf16

  dim3 gg(MTOT / 128, D_ / 128);   // 64 x 4

  gemm_kernel<float, true><<<gg, 256, 0, stream>>>(X, Wq, Qb, MTOT, D_, D_);
  gemm_kernel<float, true><<<gg, 256, 0, stream>>>(Y, Wk, Kb, MTOT, D_, D_);
  gemm_kernel<float, true><<<gg, 256, 0, stream>>>(Z, Wo, Vb, MTOT, D_, D_);

  attn_kernel<<<dim3(B_ * H_, S_ / 64), 128, 0, stream>>>(Qb, Kb, Vb, Ab);

  gemm_kernel<bf16, false><<<gg, 256, 0, stream>>>(Ab, Wo, out, MTOT, D_, D_);
}
